// AttentionModule_39616778338491
// MI455X (gfx1250) — matrix operand order repaired and checked by reading, not compiled, not run on hardware
//
#include <hip/hip_runtime.h>
#include <math.h>
#include <stdint.h>

typedef __bf16 bf16;
typedef __attribute__((ext_vector_type(16))) __bf16 v16bf;
typedef __attribute__((ext_vector_type(8)))  __bf16 v8bf;
typedef __attribute__((ext_vector_type(8)))  float  v8f;
typedef __attribute__((ext_vector_type(4)))  unsigned int u32x4;

#define BDIM   4
#define NSEQ   4096
#define EDIM   256
#define HDIM   1024
#define NW     12
#define NLINKS 13
#define ROWS   (BDIM * NSEQ)   // 16384

// CDNA5 async global->LDS copy path (GLOBAL_LOAD_ASYNC_TO_LDS_B128, ASYNCcnt).
#define USE_ASYNC_LDS 1

__device__ __forceinline__ float gelu_exact(float x) {
  return 0.5f * x * (1.0f + erff(x * 0.70710678118654752440f));
}

__device__ __forceinline__ void copy16_g2l(bf16* lds_dst, const bf16* gsrc) {
#if USE_ASYNC_LDS
  unsigned lds_off = (unsigned)(uintptr_t)lds_dst;           // low 32 bits = LDS offset
  unsigned long long ga = (unsigned long long)(uintptr_t)gsrc;
  asm volatile("global_load_async_to_lds_b128 %0, %1, off"
               :: "v"(lds_off), "v"(ga) : "memory");
#else
  *(u32x4*)lds_dst = *(const u32x4*)gsrc;
#endif
}

__device__ __forceinline__ void wait_async_copies() {
#if USE_ASYNC_LDS
  asm volatile("s_wait_asynccnt 0x0" ::: "memory");
#endif
}

// ---------------- elementwise converts (transpose weights to [N][K] bf16) ----------------
__global__ void cvt_f32_bf16(const float* __restrict__ src, bf16* __restrict__ dst, int n) {
  int i = blockIdx.x * blockDim.x + threadIdx.x;
  int stride = gridDim.x * blockDim.x;
  for (; i < n; i += stride) dst[i] = (bf16)src[i];
}

// src [batch][Kd][Nd] f32 -> dst [batch][Nd][Kd] bf16
__global__ void cvt_t_f32_bf16(const float* __restrict__ src, bf16* __restrict__ dst,
                               int Kd, int Nd, int total) {
  int i = blockIdx.x * blockDim.x + threadIdx.x;
  int stride = gridDim.x * blockDim.x;
  const int kn = Kd * Nd;
  for (; i < total; i += stride) {
    int b = i / kn, rem = i - b * kn;
    int k = rem / Nd, n = rem - k * Nd;
    dst[(size_t)b * kn + (size_t)n * Kd + k] = (bf16)src[i];
  }
}

// f_W2 [NW][HDIM][13] f32 -> [NW][16][HDIM] bf16, transposed + zero padded
__global__ void cvt_fW2_pad_t(const float* __restrict__ src, bf16* __restrict__ dst) {
  int i = blockIdx.x * blockDim.x + threadIdx.x;
  int stride = gridDim.x * blockDim.x;
  const int total = NW * 16 * HDIM;
  for (; i < total; i += stride) {
    int m = i / (16 * HDIM), rem = i - m * (16 * HDIM);
    int n = rem / HDIM, k = rem - n * HDIM;
    dst[i] = (n < NLINKS) ? (bf16)src[((size_t)m * HDIM + k) * NLINKS + n] : (bf16)0.0f;
  }
}

// ---------------- WMMA GEMM ----------------
// C[M,N] = A[M,K](bf16,row-major) @ Bt[N,K](bf16,row-major == B^T) + bias,
// optional exact-GELU, out f32 or bf16 (ldc=N). K%32==0, M%128==0, N%BN==0.
template <int BN, int WAVES_M, int WAVES_N, int WM_T, int WN_T, bool GELU, bool OUT_BF16>
__global__ __launch_bounds__(WAVES_M * WAVES_N * 32)
void wmma_gemm(const bf16* __restrict__ A, const bf16* __restrict__ Bt,
               const float* __restrict__ bias, void* __restrict__ Cout,
               int M, int N, int K, int nvalid)
{
  constexpr int BM   = 128;
  constexpr int BK   = 32;
  constexpr int LDSS = BK + 8;                 // pad: 40 bf16 = 80B rows (16B aligned)
  constexpr int NTH  = WAVES_M * WAVES_N * 32;
  constexpr int ACH  = BM * BK / 8;            // 16B chunks in A tile (512)
  constexpr int BCH  = BN * BK / 8;            // 16B chunks in B tile

  __shared__ bf16 As[2][BM][LDSS];
  __shared__ bf16 Bs[2][BN][LDSS];

  const int tid  = threadIdx.x;
  const int lane = tid & 31;
  const int wave = tid >> 5;
  const int wm   = wave / WAVES_N;
  const int wn   = wave % WAVES_N;

  const int m0 = blockIdx.y * BM;
  const int n0 = blockIdx.x * BN;

  // ---- per-thread copy descriptors (hoisted; advance by BK per tile) ----
  // A: 2 chunks/thread.  chunk idx -> row = idx>>2, col = (idx&3)*8
  const int ar0 = tid >> 2,            ac0 = (tid & 3) * 8;
  const int ar1 = (tid + NTH) >> 2,    ac1 = ((tid + NTH) & 3) * 8;
  const bf16* ag0 = A + (size_t)(m0 + ar0) * K + ac0;
  const bf16* ag1 = A + (size_t)(m0 + ar1) * K + ac1;
  // B: BCH/NTH chunks/thread (>=1) or only first BCH threads
  const int br0 = tid >> 2,            bc0 = (tid & 3) * 8;
  const int br1 = (tid + NTH) >> 2,    bc1 = ((tid + NTH) & 3) * 8;
  const bf16* bg0 = Bt + (size_t)(n0 + (br0 % BN)) * K + bc0;
  const bf16* bg1 = Bt + (size_t)(n0 + (br1 % BN)) * K + bc1;

  auto stage = [&](int buf, int kt) {
    copy16_g2l(&As[buf][ar0][ac0], ag0 + kt);
    copy16_g2l(&As[buf][ar1][ac1], ag1 + kt);
    if constexpr (BCH >= NTH) {
      copy16_g2l(&Bs[buf][br0][bc0], bg0 + kt);
      if constexpr (BCH >= 2 * NTH)
        copy16_g2l(&Bs[buf][br1][bc1], bg1 + kt);
    } else {
      if (tid < BCH) copy16_g2l(&Bs[buf][br0][bc0], bg0 + kt);
    }
  };

  v8f acc[WM_T][WN_T];
#pragma unroll
  for (int mi = 0; mi < WM_T; ++mi)
#pragma unroll
    for (int ni = 0; ni < WN_T; ++ni)
#pragma unroll
      for (int r = 0; r < 8; ++r) acc[mi][ni][r] = 0.0f;

  union Frag { v16bf v; v8bf h[2]; };

  const int arow = wm * (WM_T * 16) + (lane & 15);
  const int bcol = wn * (WN_T * 16) + (lane & 15);
  const int kbA  = (lane >> 4) * 8;            // A frag: K kbA..+7 then 16+kbA..+7
  const int kbB  = (lane >> 4) * 8;            // B frag: K kbB..+7 then 16+kbB..+7

  const int NT = K / BK;
  stage(0, 0);

  for (int t = 0; t < NT; ++t) {
    const int buf = t & 1;
    wait_async_copies();
    __syncthreads();                            // copies into buf visible; prev reads done
    if (t + 1 < NT) stage(buf ^ 1, (t + 1) * BK);

    Frag af[WM_T], bfr[WN_T];
#pragma unroll
    for (int mi = 0; mi < WM_T; ++mi) {
      af[mi].h[0] = *(const v8bf*)&As[buf][arow + mi * 16][kbA];
      af[mi].h[1] = *(const v8bf*)&As[buf][arow + mi * 16][kbA + 16];
    }
#pragma unroll
    for (int ni = 0; ni < WN_T; ++ni) {
      bfr[ni].h[0] = *(const v8bf*)&Bs[buf][bcol + ni * 16][kbB];
      bfr[ni].h[1] = *(const v8bf*)&Bs[buf][bcol + ni * 16][kbB + 16];
    }
#pragma unroll
    for (int mi = 0; mi < WM_T; ++mi)
#pragma unroll
      for (int ni = 0; ni < WN_T; ++ni)
        acc[mi][ni] = __builtin_amdgcn_wmma_f32_16x16x32_bf16(
            false, af[mi].v, false, bfr[ni].v, (short)0, acc[mi][ni], false, false);
  }

  // epilogue: C/D layout — lane=(rowhalf<<4)|col ; VGPR r holds M=r+rowhalf*8
  const int rsel = (lane >> 4) * 8;
  const int csel = lane & 15;
#pragma unroll
  for (int ni = 0; ni < WN_T; ++ni) {
    const int nn = n0 + wn * (WN_T * 16) + ni * 16 + csel;
    const float bv = (nn < nvalid) ? bias[nn] : 0.0f;
#pragma unroll
    for (int mi = 0; mi < WM_T; ++mi) {
      const int mb = m0 + wm * (WM_T * 16) + mi * 16 + rsel;
#pragma unroll
      for (int r = 0; r < 8; ++r) {
        float v = acc[mi][ni][r] + bv;
        if (GELU) v = gelu_exact(v);
        if (OUT_BF16) ((bf16 *)Cout)[(size_t)(mb + r) * N + nn] = (bf16)v;
        else          ((float*)Cout)[(size_t)(mb + r) * N + nn] = v;
      }
    }
  }
}

// ---------------- chord gather-mix ----------------
// Vout[b,n,:] = Vin[b,n,:] + sum_l Wc[b,n,l] * Vin[b, cols[n,l], :]
__global__ __launch_bounds__(64)
void mix_kernel(const float* __restrict__ Vin, const float* __restrict__ Wc,
                const int* __restrict__ cols, float* __restrict__ Vout)
{
  __shared__ float w[NLINKS];
  __shared__ int   c[NLINKS];
  const int bn = blockIdx.x;          // 0..ROWS-1
  const int b  = bn / NSEQ;
  const int n  = bn % NSEQ;
  if (threadIdx.x < NLINKS) {
    w[threadIdx.x] = Wc[bn * 16 + threadIdx.x];
    c[threadIdx.x] = cols[n * NLINKS + threadIdx.x];
  }
  __syncthreads();
  const int e = threadIdx.x * 4;      // E=256, 64 threads x float4
  float4 acc = *(const float4*)(Vin + (size_t)bn * EDIM + e);   // residual
#pragma unroll 1
  for (int l = 0; l < NLINKS; ++l) {
    const float4 g = *(const float4*)(Vin + ((size_t)b * NSEQ + c[l]) * EDIM + e);
    const float wl = w[l];
    acc.x += wl * g.x; acc.y += wl * g.y; acc.z += wl * g.z; acc.w += wl * g.w;
  }
  *(float4*)(Vout + (size_t)bn * EDIM + e) = acc;
}

// ---------------- host ----------------
extern "C" void kernel_launch(void* const* d_in, const int* in_sizes, int n_in,
                              void* d_out, int out_size, void* d_ws, size_t ws_size,
                              hipStream_t stream)
{
  const float* V    = (const float*)d_in[0];
  const float* Xin  = (const float*)d_in[1];
  const float* gW1  = (const float*)d_in[2];
  const float* gb1  = (const float*)d_in[3];
  const float* gW2  = (const float*)d_in[4];
  const float* gb2  = (const float*)d_in[5];
  const float* fW1  = (const float*)d_in[6];
  const float* fb1  = (const float*)d_in[7];
  const float* fW2  = (const float*)d_in[8];
  const float* fb2  = (const float*)d_in[9];
  const int*   cols = (const int*)d_in[10];

  char* ws = (char*)d_ws;
  size_t off = 0;
  auto take = [&](size_t bytes) -> char* {
    char* p = ws + off;
    off += (bytes + 255) & ~(size_t)255;
    return p;
  };

  bf16*  Vb    = (bf16*) take((size_t)ROWS * EDIM * 2);
  bf16*  Ib    = (bf16*) take((size_t)ROWS * EDIM * 2);
  bf16*  gW1t  = (bf16*) take((size_t)HDIM * EDIM * 2);        // [H][E]
  bf16*  gW2t  = (bf16*) take((size_t)EDIM * HDIM * 2);        // [E][H]
  bf16*  fW1t  = (bf16*) take((size_t)NW * HDIM * EDIM * 2);   // [NW][H][E]
  bf16*  fW2t  = (bf16*) take((size_t)NW * 16 * HDIM * 2);     // [NW][16][H]
  bf16*  Hid   = (bf16*) take((size_t)ROWS * HDIM * 2);
  float* Wc    = (float*)take((size_t)ROWS * 16 * 4);
  float* Vp0   = (float*)take((size_t)ROWS * EDIM * 4);
  float* Vp1   = (float*)take((size_t)ROWS * EDIM * 4);

  // one-time precision converts (weights transposed to [N][K])
  cvt_f32_bf16 <<<1024, 256, 0, stream>>>(V,   Vb, ROWS * EDIM);
  cvt_f32_bf16 <<<1024, 256, 0, stream>>>(Xin, Ib, ROWS * EDIM);
  cvt_t_f32_bf16<<<512, 256, 0, stream>>>(gW1, gW1t, EDIM, HDIM, EDIM * HDIM);
  cvt_t_f32_bf16<<<512, 256, 0, stream>>>(gW2, gW2t, HDIM, EDIM, HDIM * EDIM);
  cvt_t_f32_bf16<<<2048, 256, 0, stream>>>(fW1, fW1t, EDIM, HDIM, NW * EDIM * HDIM);
  cvt_fW2_pad_t<<<512, 256, 0, stream>>>(fW2, fW2t);

  const dim3 blk(256);

  // global MLP: V <- gelu(V @ gW1 + b1) @ gW2 + b2   (result -> Vp0, f32)
  wmma_gemm<128, 2, 4, 4, 2, true,  true ><<<dim3(HDIM / 128, ROWS / 128), blk, 0, stream>>>(
      Vb, gW1t, gb1, Hid, ROWS, HDIM, EDIM, HDIM);
  wmma_gemm<128, 2, 4, 4, 2, false, false><<<dim3(EDIM / 128, ROWS / 128), blk, 0, stream>>>(
      Hid, gW2t, gb2, Vp0, ROWS, EDIM, HDIM, EDIM);

  // chord steps: coefficient GEMMs depend only on `input`; mix is the serial part
  for (int m = 0; m < NW; ++m) {
    wmma_gemm<128, 2, 4, 4, 2, true, true><<<dim3(HDIM / 128, ROWS / 128), blk, 0, stream>>>(
        Ib, fW1t + (size_t)m * HDIM * EDIM, fb1 + m * HDIM, Hid, ROWS, HDIM, EDIM, HDIM);
    wmma_gemm<16, 8, 1, 1, 1, false, false><<<dim3(1, ROWS / 128), blk, 0, stream>>>(
        Hid, fW2t + (size_t)m * 16 * HDIM, fb2 + m * NLINKS, Wc, ROWS, 16, HDIM, NLINKS);

    const float* vin  = (m & 1) ? Vp1 : Vp0;
    float*       vout = (m == NW - 1) ? (float*)d_out : ((m & 1) ? Vp0 : Vp1);
    mix_kernel<<<ROWS, 64, 0, stream>>>(vin, Wc, cols, vout);
  }
}
